// ThresholdMoeLayer_4999341932689
// MI455X (gfx1250) — matrix operand order repaired and checked by reading, not compiled, not run on hardware
//
#include <hip/hip_runtime.h>
#include <hip/hip_bf16.h>
#include <stdint.h>

// ---- CDNA5 WMMA vector types (wave32) ----
typedef __attribute__((ext_vector_type(16))) __bf16 v16bf;
typedef __attribute__((ext_vector_type(8)))  __bf16 v8bf;
typedef __attribute__((ext_vector_type(8)))  float  v8f;

// GCC-style 4-int vector: exact pointee type of the async-copy builtin params
typedef int v4i_b128 __attribute__((vector_size(16)));
typedef __attribute__((address_space(1))) v4i_b128* g_b128_ptr;   // global
typedef __attribute__((address_space(3))) v4i_b128* l_b128_ptr;   // LDS

#define T_TOK 16384   // B*S = 4*4096 tokens
#define DIM   1024
#define NEXP  8
#define BK    32      // K step = WMMA bf16 K
#define LDA   40      // padded LDS stride (halfs): rows 16B aligned, banks spread
#define LDB   40

// Detect the gfx1250 async global->LDS copy builtin (ASYNCcnt-tracked).
#if defined(__has_builtin)
#  if __has_builtin(__builtin_amdgcn_global_load_async_to_lds_b128)
#    define MOE_ASYNC 1
#  endif
#endif
#ifndef MOE_ASYNC
#  define MOE_ASYNC 0
#endif

// 16-byte global->LDS copy: async DMA path if available, else plain b128 copy.
__device__ __forceinline__ void cp16(const unsigned short* __restrict__ g,
                                     unsigned short* __restrict__ l) {
#if MOE_ASYNC
  // flat LDS address truncates to the 32-bit LDS offset (ISA 10.2);
  // addrspace(3) pointers are 32-bit.
  __builtin_amdgcn_global_load_async_to_lds_b128(
      (g_b128_ptr)(uintptr_t)g,
      (l_b128_ptr)(unsigned)(uintptr_t)l,
      0, 0);
#else
  *(uint4*)l = *(const uint4*)g;
#endif
}

__device__ __forceinline__ void wait_async() {
#if MOE_ASYNC
#  if __has_builtin(__builtin_amdgcn_s_wait_asynccnt)
  __builtin_amdgcn_s_wait_asynccnt(0);
#  else
  asm volatile("s_wait_asynccnt 0" ::: "memory");
#  endif
#endif
}

// round-to-nearest-even f32 -> bf16 (bit trick; NaN edge irrelevant here)
__device__ __forceinline__ unsigned f2bf(float f) {
  union { float f; unsigned u; } c; c.f = f;
  unsigned r = c.u + 0x7FFFu + ((c.u >> 16) & 1u);
  return r >> 16;
}

// ---------------- gating: softmax(x @ gate_w + gate_b), threshold, renorm ---
__global__ __launch_bounds__(256)
void moe_gate_kernel(const float* __restrict__ x,
                     const float* __restrict__ gw,
                     const float* __restrict__ gb,
                     float* __restrict__ wts) {
  const int tok  = (int)((blockIdx.x * blockDim.x + threadIdx.x) >> 5); // wave/token
  const int lane = threadIdx.x & 31;
  const float* xr = x + (size_t)tok * DIM;

  float acc[NEXP];
#pragma unroll
  for (int e = 0; e < NEXP; ++e) acc[e] = 0.f;

  for (int k = lane; k < DIM; k += 32) {
    const float  xv = xr[k];
    const float4 g0 = ((const float4*)(gw + (size_t)k * NEXP))[0];
    const float4 g1 = ((const float4*)(gw + (size_t)k * NEXP))[1];
    acc[0] += xv * g0.x; acc[1] += xv * g0.y; acc[2] += xv * g0.z; acc[3] += xv * g0.w;
    acc[4] += xv * g1.x; acc[5] += xv * g1.y; acc[6] += xv * g1.z; acc[7] += xv * g1.w;
  }
#pragma unroll
  for (int off = 16; off > 0; off >>= 1)
#pragma unroll
    for (int e = 0; e < NEXP; ++e)
      acc[e] += __shfl_xor(acc[e], off, 32);

  if (lane == 0) {
    float l[NEXP], mx, s = 0.f, wsum = 0.f;
#pragma unroll
    for (int e = 0; e < NEXP; ++e) l[e] = acc[e] + gb[e];
    mx = l[0];
#pragma unroll
    for (int e = 1; e < NEXP; ++e) mx = fmaxf(mx, l[e]);
#pragma unroll
    for (int e = 0; e < NEXP; ++e) { l[e] = __expf(l[e] - mx); s += l[e]; }
    const float inv = 1.f / s;
#pragma unroll
    for (int e = 0; e < NEXP; ++e) {
      float p = l[e] * inv;
      p = (p >= 0.1f) ? p : 0.f;                  // THRESHOLD gating
      l[e] = p; wsum += p;
    }
    const float wn = (wsum == 0.f) ? 1.f : wsum;  // jnp.where(wsum==0, 1, wsum)
#pragma unroll
    for (int e = 0; e < NEXP; ++e) wts[(size_t)tok * NEXP + e] = l[e] / wn;
  }
}

// ---------------- one-shot precision/layout conversion kernels -------------
// x [T,D] f32 -> bf16, same layout. 8 elems/thread.
__global__ __launch_bounds__(256)
void cvt_x_kernel(const float* __restrict__ x, unsigned short* __restrict__ xb) {
  const size_t g = (size_t)blockIdx.x * 256 + threadIdx.x;
  const float4* p = (const float4*)x + g * 2;
  const float4 a = p[0], b = p[1];
  uint4 pk;
  pk.x = f2bf(a.x) | (f2bf(a.y) << 16);
  pk.y = f2bf(a.z) | (f2bf(a.w) << 16);
  pk.z = f2bf(b.x) | (f2bf(b.y) << 16);
  pk.w = f2bf(b.z) | (f2bf(b.w) << 16);
  ((uint4*)xb)[g] = pk;
}

// expert_w [e][k][n] f32 -> wT [e][n][k] bf16 (transposed so B-tile staging in
// the GEMM is a contiguous copy). LDS 32x32 tile transpose, coalesced I/O.
__global__ void cvt_wT_kernel(const float* __restrict__ ew,
                              unsigned short* __restrict__ wtb) {
  __shared__ float tile[32][33];
  const int e  = blockIdx.z;
  const int kb = blockIdx.y * 32;
  const int nb = blockIdx.x * 32;
  const int tx = threadIdx.x;   // 0..31
  const int ty = threadIdx.y;   // 0..7
  const float* W = ew + (size_t)e * DIM * DIM;
#pragma unroll
  for (int j = 0; j < 4; ++j)
    tile[ty + j * 8][tx] = W[(size_t)(kb + ty + j * 8) * DIM + nb + tx];
  __syncthreads();
  unsigned short* O = wtb + ((size_t)e << 20);
#pragma unroll
  for (int j = 0; j < 4; ++j)
    O[(size_t)(nb + ty + j * 8) * DIM + kb + tx] =
        (unsigned short)f2bf(tile[tx][ty + j * 8]);
}

// ---------------- fast path: bf16 WMMA GEMM with async double-buffering ----
// Block tile 128(M) x 128(N), BK=32. 8 waves, each computes a 16x128 strip
// (8 WMMA tiles). Expert loop outer; A/B tiles re-streamed from L2 (x + W
// both fit in the 192MB L2). Flattened (expert,kt) loop keeps the async
// pipeline running across expert boundaries.
__global__ __launch_bounds__(256)
void moe_expert_bf16_kernel(const unsigned short* __restrict__ xb,
                            const unsigned short* __restrict__ wtb,
                            const float* __restrict__ eb,
                            const float* __restrict__ wts,
                            float* __restrict__ out) {
  __shared__ unsigned short A_lds[2][128 * LDA];  // 2 x 10KB bf16 A tile [m][k]
  __shared__ unsigned short B_lds[2][128 * LDB];  // 2 x 10KB bf16 B tile [n][k]
  __shared__ float          w_lds[128 * NEXP];    // 4KB gate weights

  const int tid    = threadIdx.x;
  const int n0     = blockIdx.x * 128;
  const int m0     = blockIdx.y * 128;
  const int wave   = tid >> 5;
  const int lane   = tid & 31;
  const int mW     = wave * 16;                 // 8 waves * 16 rows = 128
  const int l15    = lane & 15;
  const int rowOff = (lane < 16) ? 0 : 8;       // C layout: VGPR i -> row i / i+8
  const int k0a    = (lane < 16) ? 0 : 16;      // A layout half-wave K base
  const int k0b    = (lane < 16) ? 0 : 16;      // B layout half-wave K base

  // staging slots: thread copies 16B for rows r and r+64 of each tile
  const int rT = tid >> 2;                      // 0..63
  const int c8 = (tid & 3) * 8;                 // half offset within row

  // stage this M-tile's gating weights (128 tokens x 8 experts)
#pragma unroll
  for (int i = 0; i < 4; ++i) {
    const int idx = tid + i * 256;
    w_lds[idx] = wts[(size_t)m0 * NEXP + idx];
  }

  auto issue = [&](int buf, int en, int kt) {
    const unsigned short* ga  = xb + (size_t)(m0 + rT) * DIM + kt * BK + c8;
    const unsigned short* gb2 = wtb + ((size_t)en << 20)
                              + (size_t)(n0 + rT) * DIM + kt * BK + c8;
    cp16(ga,             &A_lds[buf][rT * LDA + c8]);
    cp16(ga + 64 * DIM,  &A_lds[buf][(rT + 64) * LDA + c8]);
    cp16(gb2,            &B_lds[buf][rT * LDB + c8]);
    cp16(gb2 + 64 * DIM, &B_lds[buf][(rT + 64) * LDB + c8]);
  };

  v8f acc[8], outAcc[8];
#pragma unroll
  for (int ns = 0; ns < 8; ++ns)
#pragma unroll
    for (int i = 0; i < 8; ++i) { acc[ns][i] = 0.f; outAcc[ns][i] = 0.f; }

  issue(0, 0, 0);

  const int TOT = NEXP * (DIM / BK);            // 256 pipelined stages
#pragma unroll 2
  for (int it = 0; it < TOT; ++it) {
    const int e   = it >> 5;
    const int kt  = it & 31;
    const int cur = it & 1;

    wait_async();          // this wave's copies for tile `it` are in LDS
    __syncthreads();       // all waves' copies done; prev buffer fully read

    if (it + 1 < TOT) issue(1 - cur, (it + 1) >> 5, (it + 1) & 31);

    // ---- fragments (A and B in contiguous k = 16*(lane/16) + s order) + 8 WMMAs
    const unsigned short* Ab = A_lds[cur];
    const unsigned short* Bb = B_lds[cur];
    const int ar = (mW + l15) * LDA;
    const v8bf alo = *(const v8bf*)&Ab[ar + k0a];
    const v8bf ahi = *(const v8bf*)&Ab[ar + k0a + 8];
    v16bf a;
#pragma unroll
    for (int i = 0; i < 8; ++i) { a[i] = alo[i]; a[i + 8] = ahi[i]; }

#pragma unroll
    for (int ns = 0; ns < 8; ++ns) {
      const int br = (ns * 16 + l15) * LDB;
      const v8bf blo = *(const v8bf*)&Bb[br + k0b];
      const v8bf bhi = *(const v8bf*)&Bb[br + k0b + 8];
      v16bf b;
#pragma unroll
      for (int i = 0; i < 8; ++i) { b[i] = blo[i]; b[i + 8] = bhi[i]; }
      acc[ns] = __builtin_amdgcn_wmma_f32_16x16x32_bf16(
          false, a, false, b, (short)0, acc[ns], false, false);
    }

    // ---- per-expert epilogue: out += w[t,e] * (x@W_e + b_e)
    if (kt == 31) {
#pragma unroll
      for (int ns = 0; ns < 8; ++ns) {
        const float bias = eb[(size_t)e * DIM + n0 + ns * 16 + l15];
#pragma unroll
        for (int i = 0; i < 8; ++i) {
          const float wtok = w_lds[(mW + rowOff + i) * NEXP + e];
          outAcc[ns][i] += wtok * (acc[ns][i] + bias);
          acc[ns][i] = 0.f;
        }
      }
    }
  }

  // ---- store f32 result (C layout: VGPR i, lane half -> row i / i+8)
#pragma unroll
  for (int ns = 0; ns < 8; ++ns) {
    const int col = n0 + ns * 16 + l15;
#pragma unroll
    for (int i = 0; i < 8; ++i) {
      const int row = m0 + mW + rowOff + i;
      out[(size_t)row * DIM + col] = outAcc[ns][i];
    }
  }
}

// ---------------- fallback path (small ws): f32-source kernel --------------
__global__ __launch_bounds__(256)
void moe_expert_f32_kernel(const float* __restrict__ x,
                           const float* __restrict__ ew,
                           const float* __restrict__ eb,
                           const float* __restrict__ wts,
                           float* __restrict__ out) {
  __shared__ unsigned short A_lds[128 * LDA];
  __shared__ unsigned short B_lds[64 * LDB];
  __shared__ float          w_lds[128 * NEXP];

  const int tid    = threadIdx.x;
  const int n0     = blockIdx.x * 64;
  const int m0     = blockIdx.y * 128;
  const int wave   = tid >> 5;
  const int lane   = tid & 31;
  const int mW     = wave * 16;
  const int l15    = lane & 15;
  const int rowOff = (lane < 16) ? 0 : 8;
  const int k0a    = (lane < 16) ? 0 : 16;
  const int k0b    = (lane < 16) ? 0 : 16;

#pragma unroll
  for (int i = 0; i < 4; ++i) {
    const int idx = tid + i * 256;
    w_lds[idx] = wts[(size_t)m0 * NEXP + idx];
  }

  v8f outAcc[4];
#pragma unroll
  for (int ns = 0; ns < 4; ++ns)
#pragma unroll
    for (int i = 0; i < 8; ++i) outAcc[ns][i] = 0.f;

#pragma unroll 1
  for (int e = 0; e < NEXP; ++e) {
    const float* We = ew + (size_t)e * DIM * DIM;
    v8f acc[4];
#pragma unroll
    for (int ns = 0; ns < 4; ++ns)
#pragma unroll
      for (int i = 0; i < 8; ++i) acc[ns][i] = 0.f;

#pragma unroll 1
    for (int kt = 0; kt < DIM / BK; ++kt) {
      __syncthreads();
#pragma unroll
      for (int p = 0; p < 4; ++p) {
        const int r  = p * 32 + (tid >> 3);
        const int c4 = (tid & 7) * 4;
        const float4 v = *(const float4*)(x + (size_t)(m0 + r) * DIM + kt * BK + c4);
        uint2 pk;
        pk.x = f2bf(v.x) | (f2bf(v.y) << 16);
        pk.y = f2bf(v.z) | (f2bf(v.w) << 16);
        *(uint2*)(&A_lds[r * LDA + c4]) = pk;
      }
#pragma unroll
      for (int p = 0; p < 2; ++p) {
        const int kk = p * 16 + (tid >> 4);
        const int c4 = (tid & 15) * 4;
        const float4 v = *(const float4*)(We + (size_t)(kt * BK + kk) * DIM + n0 + c4);
        B_lds[(c4 + 0) * LDB + kk] = (unsigned short)f2bf(v.x);
        B_lds[(c4 + 1) * LDB + kk] = (unsigned short)f2bf(v.y);
        B_lds[(c4 + 2) * LDB + kk] = (unsigned short)f2bf(v.z);
        B_lds[(c4 + 3) * LDB + kk] = (unsigned short)f2bf(v.w);
      }
      __syncthreads();

      const int ar = (mW + l15) * LDA;
      const v8bf alo = *(const v8bf*)&A_lds[ar + k0a];
      const v8bf ahi = *(const v8bf*)&A_lds[ar + k0a + 8];
      v16bf a;
#pragma unroll
      for (int i = 0; i < 8; ++i) { a[i] = alo[i]; a[i + 8] = ahi[i]; }

#pragma unroll
      for (int ns = 0; ns < 4; ++ns) {
        const int br = (ns * 16 + l15) * LDB;
        const v8bf blo = *(const v8bf*)&B_lds[br + k0b];
        const v8bf bhi = *(const v8bf*)&B_lds[br + k0b + 8];
        v16bf b;
#pragma unroll
        for (int i = 0; i < 8; ++i) { b[i] = blo[i]; b[i + 8] = bhi[i]; }
        acc[ns] = __builtin_amdgcn_wmma_f32_16x16x32_bf16(
            false, a, false, b, (short)0, acc[ns], false, false);
      }
    }

#pragma unroll
    for (int ns = 0; ns < 4; ++ns) {
      const float bias = eb[(size_t)e * DIM + n0 + ns * 16 + l15];
#pragma unroll
      for (int i = 0; i < 8; ++i) {
        const float wtok = w_lds[(mW + rowOff + i) * NEXP + e];
        outAcc[ns][i] += wtok * (acc[ns][i] + bias);
      }
    }
  }

#pragma unroll
  for (int ns = 0; ns < 4; ++ns) {
    const int col = n0 + ns * 16 + l15;
#pragma unroll
    for (int i = 0; i < 8; ++i) {
      const int row = m0 + mW + rowOff + i;
      out[(size_t)row * DIM + col] = outAcc[ns][i];
    }
  }
}

extern "C" void kernel_launch(void* const* d_in, const int* in_sizes, int n_in,
                              void* d_out, int out_size, void* d_ws, size_t ws_size,
                              hipStream_t stream) {
  (void)in_sizes; (void)n_in; (void)out_size;
  const float* x   = (const float*)d_in[0];  // [4,4096,1024] fp32
  const float* gw  = (const float*)d_in[1];  // [1024,8]
  const float* gb  = (const float*)d_in[2];  // [8]
  const float* ewt = (const float*)d_in[3];  // [8,1024,1024]
  const float* ebi = (const float*)d_in[4];  // [8,1024]
  float* out = (float*)d_out;                // [16384,1024] fp32

  const size_t WTS_BYTES = (size_t)T_TOK * NEXP * sizeof(float);   // 512KB
  const size_t XB_BYTES  = (size_t)T_TOK * DIM * 2;                // 32MB
  const size_t WT_BYTES  = (size_t)NEXP * DIM * DIM * 2;           // 16MB

  float* wts = (float*)d_ws;
  moe_gate_kernel<<<dim3(T_TOK / 8), dim3(256), 0, stream>>>(x, gw, gb, wts);

  if (ws_size >= WTS_BYTES + XB_BYTES + WT_BYTES) {
    unsigned short* xb  = (unsigned short*)((char*)d_ws + WTS_BYTES);
    unsigned short* wtb = (unsigned short*)((char*)d_ws + WTS_BYTES + XB_BYTES);
    cvt_x_kernel<<<dim3((T_TOK * DIM) / (256 * 8)), dim3(256), 0, stream>>>(x, xb);
    cvt_wT_kernel<<<dim3(DIM / 32, DIM / 32, NEXP), dim3(32, 8), 0, stream>>>(ewt, wtb);
    moe_expert_bf16_kernel<<<dim3(DIM / 128, T_TOK / 128), dim3(256), 0, stream>>>(
        xb, wtb, ebi, wts, out);
  } else {
    moe_expert_f32_kernel<<<dim3(DIM / 64, T_TOK / 128), dim3(256), 0, stream>>>(
        x, ewt, ebi, wts, out);
  }
}
